// LocalAttention_12438225289574
// MI455X (gfx1250) — matrix operand order repaired and checked by reading, not compiled, not run on hardware
//
#include <hip/hip_runtime.h>
#include <hip/hip_bf16.h>

// ---------------------------------------------------------------------------
// Local attention for MI455X (gfx1250, wave32, WMMA bf16 16x16x32, f32 accum)
// Register-blocked: each wave computes a 16x64 output tile (4 WMMAs / A-frag)
// ---------------------------------------------------------------------------

typedef __attribute__((ext_vector_type(16))) __bf16 v16bf;
typedef __attribute__((ext_vector_type(8)))  __bf16 bf16x8;
typedef __attribute__((ext_vector_type(8)))  float  v8f;
typedef __attribute__((ext_vector_type(8)))  float  f32x8;

constexpr int B_ = 8;
constexpr int L_ = 2048;
constexpr int E_ = 512;
constexpr int HALF_W = 128;   // window_size / 2

// ---- ordered-uint encoding for float atomic max in LDS --------------------
__device__ __forceinline__ unsigned ford(float f) {
    unsigned u = __float_as_uint(f);
    return (u & 0x80000000u) ? ~u : (u | 0x80000000u);
}
__device__ __forceinline__ float funord(unsigned s) {
    unsigned u = (s & 0x80000000u) ? (s ^ 0x80000000u) : ~s;
    return __uint_as_float(u);
}

// ---- WMMA wrapper ---------------------------------------------------------
__device__ __forceinline__ v8f wmma_bf16(v16bf a, v16bf b, v8f c) {
    // (neg_a, A, neg_b, B, c_mod, C, reuse_a, reuse_b)
    return __builtin_amdgcn_wmma_f32_16x16x32_bf16(false, a, false, b,
                                                   (short)0, c, false, false);
}

// ---- fragment loaders -----------------------------------------------------
// A-matrix 16x32 bf16 layout (ISA 7.12.2): lanes 0-15 row M=lane, K 0-7 &
// 16-23; lanes 16-31 row M=lane-16, K 8-15 & 24-31.
__device__ __forceinline__ v16bf frag_a_f32(const float* __restrict__ p, int ld, int lane) {
    const float* r = p + (size_t)(lane & 15) * ld;
    int hk = (lane >> 4) & 1;
    f32x8 lo8 = *reinterpret_cast<const f32x8*>(r + hk * 8);
    f32x8 hi8 = *reinterpret_cast<const f32x8*>(r + 16 + hk * 8);
    v16bf a;
#pragma unroll
    for (int j = 0; j < 8; ++j) { a[j] = (__bf16)lo8[j]; a[8 + j] = (__bf16)hi8[j]; }
    return a;
}
__device__ __forceinline__ v16bf frag_a_bf16(const __bf16* __restrict__ p, int ld, int lane) {
    const __bf16* r = p + (size_t)(lane & 15) * ld;
    int hk = (lane >> 4) & 1;
    bf16x8 lo8 = *reinterpret_cast<const bf16x8*>(r + hk * 8);
    bf16x8 hi8 = *reinterpret_cast<const bf16x8*>(r + 16 + hk * 8);
    v16bf a;
#pragma unroll
    for (int j = 0; j < 8; ++j) { a[j] = lo8[j]; a[8 + j] = hi8[j]; }
    return a;
}
// B-matrix 32x16: lane holds column n = lane&15, K = hk*8 + 0..7 & 16 + hk*8 + 0..7
// (hk = lane>>4), matching A (source is B^T row-major: row n = column n).
__device__ __forceinline__ v16bf frag_b_bf16(const __bf16* __restrict__ p, int ld, int lane) {
    const __bf16* r = p + (size_t)(lane & 15) * ld;
    int hk = (lane >> 4) & 1;
    bf16x8 lo8 = *reinterpret_cast<const bf16x8*>(r + hk * 8);
    bf16x8 hi8 = *reinterpret_cast<const bf16x8*>(r + 16 + hk * 8);
    v16bf b;
#pragma unroll
    for (int j = 0; j < 8; ++j) { b[j] = lo8[j]; b[8 + j] = hi8[j]; }
    return b;
}

// ---------------------------------------------------------------------------
// Kernel 0: one-shot f32 -> bf16 conversion of Wq|Wk|Wv into Wb[3, E, E].
// ---------------------------------------------------------------------------
__global__ void convert_w_kernel(const float* __restrict__ Wq,
                                 const float* __restrict__ Wk,
                                 const float* __restrict__ Wv,
                                 __bf16* __restrict__ Wb) {
    constexpr int N = E_ * E_;
    int idx = blockIdx.x * blockDim.x + threadIdx.x;
    if (idx >= 3 * N) return;
    const float* src = (idx < N) ? Wq : (idx < 2 * N) ? Wk : Wv;
    int off = (idx < N) ? idx : (idx < 2 * N) ? (idx - N) : (idx - 2 * N);
    Wb[idx] = (__bf16)src[off];
}

// ---------------------------------------------------------------------------
// Kernel 1: Q/K/V projection.  y = x @ W^T + b, output bf16.
// grid: (B*L/128, E/64, 3); block 256 (8 waves, each one 16x64 tile).
// z=0 -> Qb[b*L+l, f], z=1 -> Kb, z=2 -> Vt[b, f, l] (transposed).
// ---------------------------------------------------------------------------
__global__ void qkv_proj_kernel(const float* __restrict__ x,
                                const __bf16* __restrict__ Wb,
                                const float* __restrict__ bq,
                                const float* __restrict__ bk,
                                const float* __restrict__ bv,
                                __bf16* __restrict__ Qb, __bf16* __restrict__ Kb,
                                __bf16* __restrict__ Vt) {
    const int lane = threadIdx.x & 31;
    const int wave = threadIdx.x >> 5;
    const int row0 = blockIdx.x * 128 + wave * 16;   // global row in [0, B*L)
    const int n0   = blockIdx.y * 64;                // output feature tile base
    const int which = blockIdx.z;

    const __bf16* W    = Wb + (size_t)which * E_ * E_;
    const float*  bias = (which == 0) ? bq : (which == 1) ? bk : bv;

    v8f acc[4];
#pragma unroll
    for (int j = 0; j < 4; ++j) { v8f z = {}; acc[j] = z; }

    for (int k = 0; k < E_; k += 32) {
        v16bf a = frag_a_f32(x + (size_t)row0 * E_ + k, E_, lane);
#pragma unroll
        for (int j = 0; j < 4; ++j) {
            v16bf bfrag = frag_b_bf16(W + (size_t)(n0 + j * 16) * E_ + k, E_, lane);
            acc[j] = wmma_bf16(a, bfrag, acc[j]);
        }
    }

    const int nloc  = lane & 15;
    const int mbase = row0 + 8 * ((lane >> 4) & 1);

    if (which == 2) {
        // store V transposed: Vt[b, f, l]
        const int bi = row0 >> 11;                 // / L (tiles never cross b)
        __bf16* vt = Vt + (size_t)bi * E_ * L_;
#pragma unroll
        for (int j = 0; j < 4; ++j) {
            int n = n0 + j * 16 + nloc;
            float bb = bias[n];
#pragma unroll
            for (int r = 0; r < 8; ++r) {
                int l = (mbase + r) & (L_ - 1);
                vt[(size_t)n * L_ + l] = (__bf16)(acc[j][r] + bb);
            }
        }
    } else {
        __bf16* dst = (which == 0) ? Qb : Kb;
#pragma unroll
        for (int j = 0; j < 4; ++j) {
            int n = n0 + j * 16 + nloc;
            float bb = bias[n];
#pragma unroll
            for (int r = 0; r < 8; ++r)
                dst[(size_t)(mbase + r) * E_ + n] = (__bf16)(acc[j][r] + bb);
        }
    }
}

// ---------------------------------------------------------------------------
// Kernel 2: banded scores (WMMA) + full-row softmax with analytic background.
// grid: (L/16, B); block 256. Each wave handles up to 3 of <=17 key tiles.
// Writes attn f32 to d_out and (optionally) bf16 copy for the A*V GEMM.
// ---------------------------------------------------------------------------
__global__ void attn_softmax_kernel(const __bf16* __restrict__ Qb,
                                    const __bf16* __restrict__ Kb,
                                    float* __restrict__ attn,
                                    __bf16* __restrict__ attnb /* may be null */) {
    const int lane = threadIdx.x & 31;
    const int wave = threadIdx.x >> 5;
    const int b  = blockIdx.y;
    const int q0 = blockIdx.x * 16;

    int lo  = q0 - HALF_W;      if (lo < 0)   lo = 0;
    int end = q0 + 16 + HALF_W; if (end > L_) end = L_;
    const int ntiles = (end - lo) >> 4;      // <= 17

    __shared__ unsigned s_max[16];
    __shared__ float    s_sum[16];
    __shared__ float    s_bg[16];
    if (threadIdx.x < 16) { s_max[threadIdx.x] = ford(0.0f); s_sum[threadIdx.x] = 0.0f; }
    __syncthreads();

    const __bf16* Qbase = Qb + ((size_t)b * L_ + q0) * E_;
    const __bf16* Kbase = Kb + (size_t)b * L_ * E_;
    const float scale = 0.044194173824159216f;  // 1/sqrt(512)

    v8f  sc[3];
    int  c0s[3];
    bool act[3];
#pragma unroll
    for (int i = 0; i < 3; ++i) {
        int t = wave + 8 * i;
        act[i] = (t < ntiles);
        c0s[i] = lo + t * 16;
        v8f z = {};
        sc[i] = z;
    }
    for (int k = 0; k < E_; k += 32) {
        v16bf a = frag_a_bf16(Qbase + k, E_, lane);
#pragma unroll
        for (int i = 0; i < 3; ++i) {
            if (act[i]) {   // wave-uniform branch: EXEC stays all-ones for WMMA
                v16bf bfrag = frag_b_bf16(Kbase + (size_t)c0s[i] * E_ + k, E_, lane);
                sc[i] = wmma_bf16(a, bfrag, sc[i]);
            }
        }
    }

    const int nloc = lane & 15;
    const int hi   = (lane >> 4) & 1;

    // scale + multiplicative band mask, then per-row max via LDS atomics
#pragma unroll
    for (int i = 0; i < 3; ++i) {
        if (!act[i]) continue;
        int col = c0s[i] + nloc;
#pragma unroll
        for (int r = 0; r < 8; ++r) {
            int mrow = r + 8 * hi;
            int qg = q0 + mrow;
            float v = sc[i][r] * scale;
            int d = col - qg; if (d < 0) d = -d;
            if (d > HALF_W) v = 0.0f;        // multiplicative mask (NOT -inf)
            sc[i][r] = v;
            atomicMax(&s_max[mrow], ford(v));
        }
    }
    __syncthreads();

    // exp + per-row sum
#pragma unroll
    for (int i = 0; i < 3; ++i) {
        if (!act[i]) continue;
#pragma unroll
        for (int r = 0; r < 8; ++r) {
            int mrow = r + 8 * hi;
            float mx = funord(s_max[mrow]);
            float e = __expf(sc[i][r] - mx);
            sc[i][r] = e;
            atomicAdd(&s_sum[mrow], e);
        }
    }
    // out-of-band columns all carry score 0 -> contribute (L - band)*exp(-max)
    if (threadIdx.x < 16) {
        int qg = q0 + (int)threadIdx.x;
        int jlo = qg - HALF_W;  if (jlo < 0)      jlo = 0;
        int jhi = qg + HALF_W;  if (jhi > L_ - 1) jhi = L_ - 1;
        int inband = jhi - jlo + 1;
        float mx = funord(s_max[threadIdx.x]);
        atomicAdd(&s_sum[threadIdx.x], (float)(L_ - inband) * __expf(-mx));
    }
    __syncthreads();

    if (threadIdx.x < 16) {
        float mx = funord(s_max[threadIdx.x]);
        s_bg[threadIdx.x] = __expf(-mx) / s_sum[threadIdx.x];
    }

    // write in-band attention weights (f32 output + optional bf16 copy)
    float*  arow = attn + (size_t)b * L_ * L_ + (size_t)q0 * L_;
    __bf16* brow = attnb ? attnb + (size_t)b * L_ * L_ + (size_t)q0 * L_ : nullptr;
#pragma unroll
    for (int i = 0; i < 3; ++i) {
        if (!act[i]) continue;
        int col = c0s[i] + nloc;
#pragma unroll
        for (int r = 0; r < 8; ++r) {
            int mrow = r + 8 * hi;
            float w = sc[i][r] / s_sum[mrow];
            arow[(size_t)mrow * L_ + col] = w;
            if (brow) brow[(size_t)mrow * L_ + col] = (__bf16)w;
        }
    }
    __syncthreads();   // s_bg visible to all threads

    // background fill for the remaining 2048-wide row
    for (int idx = threadIdx.x; idx < 16 * L_; idx += blockDim.x) {
        int mrow = idx >> 11;          // / 2048
        int col  = idx & (L_ - 1);
        if (col < lo || col >= end) {
            float w = s_bg[mrow];
            arow[(size_t)mrow * L_ + col] = w;
            if (brow) brow[(size_t)mrow * L_ + col] = (__bf16)w;
        }
    }
}

// ---------------------------------------------------------------------------
// Kernel 3: output = attn @ V.  Full K=2048 contraction (background mass is
// significant).  ABF16: read bf16 attn copy from ws; else f32 from d_out.
// grid: (B*L/128, E/64); block 256 (8 waves, each one 16x64 tile).
// ---------------------------------------------------------------------------
template <bool ABF16>
__global__ void av_gemm_kernel(const float* __restrict__ attnf,
                               const __bf16* __restrict__ attnb,
                               const __bf16* __restrict__ Vt,
                               float* __restrict__ out) {
    const int lane = threadIdx.x & 31;
    const int wave = threadIdx.x >> 5;
    const int row0 = blockIdx.x * 128 + wave * 16;   // global (b*L + l)
    const int n0   = blockIdx.y * 64;
    const int b  = row0 >> 11;
    const int l0 = row0 & (L_ - 1);

    const float*  arowf = attnf + (size_t)b * L_ * L_ + (size_t)l0 * L_;
    const __bf16* arowb = attnb + (size_t)b * L_ * L_ + (size_t)l0 * L_;
    const __bf16* vb    = Vt + (size_t)b * E_ * L_;

    v8f acc[4];
#pragma unroll
    for (int j = 0; j < 4; ++j) { v8f z = {}; acc[j] = z; }

    for (int k = 0; k < L_; k += 32) {
        v16bf a;
        if constexpr (ABF16) a = frag_a_bf16(arowb + k, L_, lane);
        else                 a = frag_a_f32(arowf + k, L_, lane);
#pragma unroll
        for (int j = 0; j < 4; ++j) {
            v16bf bfrag = frag_b_bf16(vb + (size_t)(n0 + j * 16) * L_ + k, L_, lane);
            acc[j] = wmma_bf16(a, bfrag, acc[j]);
        }
    }

    const int nloc  = lane & 15;
    const int mbase = row0 + 8 * ((lane >> 4) & 1);
#pragma unroll
    for (int j = 0; j < 4; ++j) {
        int n = n0 + j * 16 + nloc;
#pragma unroll
        for (int r = 0; r < 8; ++r)
            out[(size_t)(mbase + r) * E_ + n] = acc[j][r];
    }
}

// ---------------------------------------------------------------------------
// Kernel 4: pooled = mean over L of output.
// ---------------------------------------------------------------------------
__global__ void pool_kernel(const float* __restrict__ out, float* __restrict__ pooled) {
    int idx = blockIdx.x * blockDim.x + threadIdx.x;   // b*E + e
    if (idx >= B_ * E_) return;
    int b = idx / E_, e = idx % E_;
    const float* p = out + (size_t)b * L_ * E_ + e;
    float s = 0.0f;
    for (int l = 0; l < L_; ++l) s += p[(size_t)l * E_];
    pooled[idx] = s * (1.0f / (float)L_);
}

// ---------------------------------------------------------------------------
extern "C" void kernel_launch(void* const* d_in, const int* in_sizes, int n_in,
                              void* d_out, int out_size, void* d_ws, size_t ws_size,
                              hipStream_t stream) {
    const float* x  = (const float*)d_in[0];
    const float* Wq = (const float*)d_in[1];
    const float* bq = (const float*)d_in[2];
    const float* Wk = (const float*)d_in[3];
    const float* bk = (const float*)d_in[4];
    const float* Wv = (const float*)d_in[5];
    const float* bv = (const float*)d_in[6];

    float* out    = (float*)d_out;                       // [B, L, E]
    float* attn   = out + (size_t)B_ * L_ * E_;          // [B, L, L]
    float* pooled = attn + (size_t)B_ * L_ * L_;         // [B, E]

    const size_t nQ = (size_t)B_ * L_ * E_;              // per-tensor bf16 elems
    const size_t nW = (size_t)3 * E_ * E_;
    const size_t nA = (size_t)B_ * L_ * L_;

    __bf16* Qb = (__bf16*)d_ws;                          // [B*L, E]
    __bf16* Kb = Qb + nQ;                                // [B*L, E]
    __bf16* Vt = Kb + nQ;                                // [B, E, L]
    __bf16* Wb = Vt + nQ;                                // [3, E, E]
    const size_t baseBytes = (3 * nQ + nW) * sizeof(__bf16);
    const size_t fullBytes = baseBytes + nA * sizeof(__bf16);
    __bf16* Ab = (ws_size >= fullBytes) ? (Wb + nW) : nullptr;  // [B, L, L] bf16

    convert_w_kernel<<<(int)((nW + 255) / 256), 256, 0, stream>>>(Wq, Wk, Wv, Wb);

    dim3 g1((B_ * L_) / 128, E_ / 64, 3);
    qkv_proj_kernel<<<g1, 256, 0, stream>>>(x, Wb, bq, bk, bv, Qb, Kb, Vt);

    dim3 g2(L_ / 16, B_);
    attn_softmax_kernel<<<g2, 256, 0, stream>>>(Qb, Kb, attn, Ab);

    dim3 g3((B_ * L_) / 128, E_ / 64);
    if (Ab) av_gemm_kernel<true ><<<g3, 256, 0, stream>>>(attn, Ab, Vt, out);
    else    av_gemm_kernel<false><<<g3, 256, 0, stream>>>(attn, Ab, Vt, out);

    pool_kernel<<<(B_ * E_ + 255) / 256, 256, 0, stream>>>(out, pooled);
}
